// GatedGCN_layer_13022340842268
// MI455X (gfx1250) — matrix operand order repaired and checked by reading, not compiled, not run on hardware
//
#include <hip/hip_runtime.h>
#include <hip/hip_bf16.h>
#include <math.h>

typedef __attribute__((ext_vector_type(16))) _Float16 v16h;
typedef __attribute__((ext_vector_type(8)))  float    v8f;
typedef int v4i __attribute__((vector_size(16)));

#define D 128
#define BN_EPS 1e-5f

#if defined(__has_builtin)
#if __has_builtin(__builtin_amdgcn_global_load_async_to_lds_b128)
#define ASYNC_LDS_COPY 1
#endif
#endif

// ---------------------------------------------------------------------------
// Kernel 0: one-shot weight prep. Convert W_A/B/D/E (fp32, [128x128] row major,
// d_in x d_out) into hi/lo f16 arrays in exact V_WMMA_F32_16X16X32_F16
// B-fragment order:  half index t = (((mtx*4+kt)*8+ct)*32+lane)*16 + h
// B layout (this code's contiguous k order): lane l -> col n=l&15 ; lanes 0-15 K=kbase+0..15,
// lanes 16-31 K=kbase+16..31 (contiguous in h).
// ---------------------------------------------------------------------------
__global__ __launch_bounds__(256) void wprep_kernel(
    const float* __restrict__ WA, const float* __restrict__ WB,
    const float* __restrict__ WD, const float* __restrict__ WE,
    _Float16* __restrict__ whi, _Float16* __restrict__ wlo) {
  const int t = blockIdx.x * 256 + threadIdx.x;  // 65536 total
  const int h    = t & 15;
  const int lane = (t >> 4) & 31;
  const int ct   = (t >> 9) & 7;
  const int kt   = (t >> 12) & 3;
  const int mtx  = t >> 14;
  const float* W = (mtx == 0) ? WA : (mtx == 1) ? WB : (mtx == 2) ? WD : WE;
  const int nn = lane & 15;
  const int k  = kt * 32 + ((lane < 16) ? 0 : 16) + h;
  const float w = W[(size_t)k * D + ct * 16 + nn];
  const _Float16 hi = (_Float16)w;
  whi[t] = hi;
  wlo[t] = (_Float16)(w - (float)hi);
}

// ---------------------------------------------------------------------------
// Kernel 1: fused 4-way projection GEMM. Block = 256 threads = 8 waves.
// Stage X tile (async DMA to LDS when available), convert ONCE to
// fragment-ordered hi/lo f16 in LDS, then each wave runs a pure
// DS-load / global-v16h-load / WMMA inner loop.
// A layout (same contiguous k order as B): lane l -> row m=l&15 ; lanes 0-15 hold K halves
// 0..15, lanes 16-31 hold 16..31 within each K=32 step.
// ---------------------------------------------------------------------------
__global__ __launch_bounds__(256) void gemm4_kernel(
    const float* __restrict__ X,
    const _Float16* __restrict__ whi, const _Float16* __restrict__ wlo,
    const float* __restrict__ bA, const float* __restrict__ bB,
    const float* __restrict__ bD, const float* __restrict__ bE,
    float* __restrict__ AX, float* __restrict__ BX,
    float* __restrict__ DX, float* __restrict__ EX, int n) {
  __shared__ __align__(32) float    xs[16 * D];        // 8 KB staging
  __shared__ __align__(32) _Float16 ahi[4 * 32 * 16];  // 4 KB frag-ordered
  __shared__ __align__(32) _Float16 alo[4 * 32 * 16];  // 4 KB frag-ordered

  const int m0 = blockIdx.x * 16;

  // speculative next-tile prefetch
  if (threadIdx.x == 0)
    __builtin_prefetch(X + (size_t)(m0 + 16) * D, 0, 1);

  // ---- stage tile: 512 float4 by 256 threads (async DMA to LDS if possible)
  {
    const float4* gsrc = (const float4*)(X + (size_t)m0 * D);
    float4* lds4 = (float4*)xs;
    for (int i = threadIdx.x; i < 512; i += 256) {
      const int row = m0 + (i >> 5);  // 32 float4 per row
      if (row < n) {
#ifdef ASYNC_LDS_COPY
        __builtin_amdgcn_global_load_async_to_lds_b128(
            (v4i __attribute__((address_space(1)))*)(gsrc + i),
            (v4i __attribute__((address_space(3)))*)(lds4 + i), 0, 0);
#else
        lds4[i] = gsrc[i];
#endif
      } else {
        lds4[i] = make_float4(0.f, 0.f, 0.f, 0.f);
      }
    }
  }
#ifdef ASYNC_LDS_COPY
#if __has_builtin(__builtin_amdgcn_s_wait_asynccnt)
  __builtin_amdgcn_s_wait_asynccnt(0);
#else
  asm volatile("s_wait_asynccnt 0x0" ::: "memory");
#endif
#endif
  __syncthreads();

  // ---- convert tile to fragment order ONCE (8 values per thread)
  for (int p = threadIdx.x; p < 2048; p += 256) {
    const int h    = p & 15;
    const int lane = (p >> 4) & 31;
    const int kt   = p >> 9;
    const int m = lane & 15;
    const int q = h >> 1, sub = h & 1;
    const int k = kt * 32 + ((lane < 16) ? 0 : 16) + h;
    const float x = xs[m * D + k];
    const _Float16 hi = (_Float16)x;
    ahi[p] = hi;
    alo[p] = (_Float16)(x - (float)hi);
  }
  __syncthreads();

  const int wave = threadIdx.x >> 5;  // 0..7 -> owns output cols [16w,16w+16)
  const int lane = threadIdx.x & 31;

  const v16h* Ah  = (const v16h*)ahi;
  const v16h* Al  = (const v16h*)alo;
  const v16h* Bh  = (const v16h*)whi;
  const v16h* Bl  = (const v16h*)wlo;

  v8f acc[4] = {};

#pragma unroll
  for (int kt = 0; kt < 4; ++kt) {
    const v16h a_hi = Ah[kt * 32 + lane];  // one DS v16h load
    const v16h a_lo = Al[kt * 32 + lane];
#pragma unroll
    for (int mtx = 0; mtx < 4; ++mtx) {
      const int fi = ((mtx * 4 + kt) * 8 + wave) * 32 + lane;
      const v16h b_hi = Bh[fi];
      const v16h b_lo = Bl[fi];
      acc[mtx] = __builtin_amdgcn_wmma_f32_16x16x32_f16(
          false, a_hi, false, b_hi, (short)0, acc[mtx], false, false);
      acc[mtx] = __builtin_amdgcn_wmma_f32_16x16x32_f16(
          false, a_hi, false, b_lo, (short)0, acc[mtx], false, false);
      acc[mtx] = __builtin_amdgcn_wmma_f32_16x16x32_f16(
          false, a_lo, false, b_hi, (short)0, acc[mtx], false, false);
    }
  }

  // ---- bias + store. C/D layout: lane -> N=l&15 ; VGPR r -> M=r+(l<16?0:8)
  const int cb = wave * 16;
  const int nn = lane & 15;
  const int mbase = (lane < 16) ? 0 : 8;
  const float* bs[4] = {bA, bB, bD, bE};
  float* Os[4] = {AX, BX, DX, EX};
#pragma unroll
  for (int mtx = 0; mtx < 4; ++mtx) {
    const float bias = bs[mtx][cb + nn];
    float* O = Os[mtx];
#pragma unroll
    for (int r = 0; r < 8; ++r) {
      const int row = m0 + mbase + r;
      if (row < n) O[(size_t)row * D + cb + nn] = acc[mtx][r] + bias;
    }
  }
}

// ---------------------------------------------------------------------------
// Kernel 2: edge gather / gate / scatter-add.  One wave per edge, float4/lane.
// BX/DX/EX (77 MB) + num/den (51 MB) stay resident in MI455X's 192 MB L2,
// so the random gathers and f32 atomic RMWs run at L2 speed, not HBM.
// ---------------------------------------------------------------------------
__global__ __launch_bounds__(256) void edge_kernel(
    const int* __restrict__ src, const int* __restrict__ dst,
    const float* __restrict__ BX, const float* __restrict__ DXp,
    const float* __restrict__ EXp,
    float* __restrict__ num, float* __restrict__ den,
    float* __restrict__ deg, int nE) {
  const int e = blockIdx.x * 8 + (threadIdx.x >> 5);
  if (e >= nE) return;
  const int lane = threadIdx.x & 31;
  const int s = src[e];
  const int d = dst[e];

  const float4 dv = ((const float4*)(DXp + (size_t)s * D))[lane];
  const float4 ev = ((const float4*)(EXp + (size_t)d * D))[lane];
  const float4 bv = ((const float4*)(BX + (size_t)s * D))[lane];

  float4 sg;
  sg.x = 1.0f / (1.0f + __expf(-(dv.x + ev.x)));
  sg.y = 1.0f / (1.0f + __expf(-(dv.y + ev.y)));
  sg.z = 1.0f / (1.0f + __expf(-(dv.z + ev.z)));
  sg.w = 1.0f / (1.0f + __expf(-(dv.w + ev.w)));

  float* np = num + (size_t)d * D + lane * 4;
  float* dp = den + (size_t)d * D + lane * 4;
  atomicAdd(np + 0, sg.x * bv.x);
  atomicAdd(np + 1, sg.y * bv.y);
  atomicAdd(np + 2, sg.z * bv.z);
  atomicAdd(np + 3, sg.w * bv.w);
  atomicAdd(dp + 0, sg.x);
  atomicAdd(dp + 1, sg.y);
  atomicAdd(dp + 2, sg.z);
  atomicAdd(dp + 3, sg.w);
  if (lane == 0) atomicAdd(deg + d, 1.0f);
}

// ---------------------------------------------------------------------------
// Kernel 3: combine + snorm scale + per-feature sum / sumsq (BN stats pass).
// ---------------------------------------------------------------------------
__global__ __launch_bounds__(256) void combine_kernel(
    const float* __restrict__ X, const float* __restrict__ AX,
    const float* __restrict__ num, const float* __restrict__ den,
    const float* __restrict__ deg, float* __restrict__ H,
    float* __restrict__ stats, int n) {
  const int f = threadIdx.x & 127;
  const int rsub = threadIdx.x >> 7;  // 0 or 1
  const int row0 = blockIdx.x * 64;
  const float inv_n = 1.0f / (float)n;

  float sum = 0.f, sumsq = 0.f;
#pragma unroll 4
  for (int i = 0; i < 32; ++i) {
    const int r = row0 + rsub + i * 2;
    if (r < n) {
      const size_t idx = (size_t)r * D + f;
      float h;
      if (deg[r] > 0.f)
        h = AX[idx] + num[idx] / den[idx];  // den > 0 whenever deg > 0
      else
        h = X[idx];
      h *= inv_n;
      H[idx] = h;
      sum += h;
      sumsq += h * h;
    }
  }
  __shared__ float ssum[256];
  __shared__ float ssq[256];
  ssum[threadIdx.x] = sum;
  ssq[threadIdx.x] = sumsq;
  __syncthreads();
  if (rsub == 0) {
    atomicAdd(&stats[f], ssum[threadIdx.x] + ssum[threadIdx.x + 128]);
    atomicAdd(&stats[128 + f], ssq[threadIdx.x] + ssq[threadIdx.x + 128]);
  }
}

// ---------------------------------------------------------------------------
// Kernel 4: fold BN stats into per-feature scale/shift.  1 block, 128 threads.
// ---------------------------------------------------------------------------
__global__ void bnscale_kernel(const float* __restrict__ stats,
                               const float* __restrict__ gamma,
                               const float* __restrict__ beta,
                               float* __restrict__ scsh, int n) {
  const int f = threadIdx.x;
  const float inv_n = 1.0f / (float)n;
  const float mean = stats[f] * inv_n;
  const float var = stats[128 + f] * inv_n - mean * mean;
  const float sc = gamma[f] * rsqrtf(var + BN_EPS);
  scsh[f] = sc;
  scsh[128 + f] = beta[f] - mean * sc;
}

// ---------------------------------------------------------------------------
// Kernel 5: affine + ReLU + residual  ->  d_out
// ---------------------------------------------------------------------------
__global__ __launch_bounds__(256) void final_kernel(
    const float* __restrict__ X, const float* __restrict__ H,
    const float* __restrict__ scsh, float* __restrict__ out, size_t total) {
  const size_t i = (size_t)blockIdx.x * 256 + threadIdx.x;
  if (i >= total) return;
  const int f = (int)(i & 127);
  float h = H[i] * scsh[f] + scsh[128 + f];
  h = fmaxf(h, 0.f);
  out[i] = X[i] + h;
}

// ---------------------------------------------------------------------------
// Host launcher
// ---------------------------------------------------------------------------
extern "C" void kernel_launch(void* const* d_in, const int* in_sizes, int n_in,
                              void* d_out, int out_size, void* d_ws,
                              size_t ws_size, hipStream_t stream) {
  const float* X     = (const float*)d_in[0];
  const float* WA    = (const float*)d_in[1];
  const float* bA    = (const float*)d_in[2];
  const float* WB    = (const float*)d_in[3];
  const float* bB    = (const float*)d_in[4];
  const float* WD    = (const float*)d_in[5];
  const float* bD    = (const float*)d_in[6];
  const float* WE    = (const float*)d_in[7];
  const float* bE    = (const float*)d_in[8];
  const float* gamma = (const float*)d_in[9];
  const float* beta  = (const float*)d_in[10];
  const int* src     = (const int*)d_in[11];
  const int* dst     = (const int*)d_in[12];

  const int n  = in_sizes[0] / D;  // 50000
  const int nE = in_sizes[11];     // 800000
  const size_t nd = (size_t)n * D;

  float* ws    = (float*)d_ws;
  float* AX    = ws;            // [n*D]
  float* BX    = AX + nd;       // [n*D]
  float* DX    = BX + nd;       // [n*D]   (reused as H after edge pass)
  float* EX    = DX + nd;       // [n*D]
  float* num   = EX + nd;       // [n*D]
  float* den   = num + nd;      // [n*D]
  float* deg   = den + nd;      // [n]
  float* stats = deg + n;       // [256]  (sum | sumsq)
  float* scsh  = stats + 256;   // [256]  (scale | shift)
  _Float16* whi = (_Float16*)(scsh + 256);  // [65536] B-frag-ordered hi
  _Float16* wlo = whi + 65536;              // [65536] B-frag-ordered lo
  float* H = DX;

  // zero the accumulators (num & den contiguous; deg & stats contiguous)
  (void)hipMemsetAsync(num, 0, 2 * nd * sizeof(float), stream);
  (void)hipMemsetAsync(deg, 0, (size_t)(n + 256) * sizeof(float), stream);

  wprep_kernel<<<256, 256, 0, stream>>>(WA, WB, WD, WE, whi, wlo);

  gemm4_kernel<<<(n + 15) / 16, 256, 0, stream>>>(
      X, whi, wlo, bA, bB, bD, bE, AX, BX, DX, EX, n);

  edge_kernel<<<(nE + 7) / 8, 256, 0, stream>>>(src, dst, BX, DX, EX, num,
                                                den, deg, nE);

  combine_kernel<<<(n + 63) / 64, 256, 0, stream>>>(X, AX, num, den, deg, H,
                                                    stats, n);

  bnscale_kernel<<<1, 128, 0, stream>>>(stats, gamma, beta, scsh, n);

  final_kernel<<<(int)((nd + 255) / 256), 256, 0, stream>>>(
      X, H, scsh, (float*)d_out, nd);
}
